// CLAM_SB_10307921510688
// MI455X (gfx1250) — matrix operand order repaired and checked by reading, not compiled, not run on hardware
//
#include <hip/hip_runtime.h>
#include <hip/hip_bf16.h>

#define BAGS 8
#define NPTS 2048
#define DIN  1024
#define HID  512
#define DATT 256
#define KNNK 5
#define NCLS 2
#define MROWS (BAGS * NPTS)   // 16384

typedef __attribute__((ext_vector_type(16))) __bf16          v16bf;
typedef __attribute__((ext_vector_type(8)))  float           v8f;
typedef __attribute__((ext_vector_type(16))) unsigned short  v16u;
typedef __attribute__((ext_vector_type(8)))  unsigned short  v8u;
typedef __attribute__((__vector_size__(4 * sizeof(int)))) int v4i_t;

union ABu { v16u u; v16bf b; };

__device__ __forceinline__ float bf2f(unsigned short u) {
  unsigned int x = ((unsigned int)u) << 16;
  float f; __builtin_memcpy(&f, &x, 4); return f;
}
__device__ __forceinline__ unsigned short f2bf(float f) {
  unsigned int x; __builtin_memcpy(&x, &f, 4);
  unsigned int r = (x + 0x7FFFu + ((x >> 16) & 1u)) >> 16;   // RNE
  return (unsigned short)r;
}

// A fragment (16x32 bf16): lane = M (%16), hi = lane>>4 selects K-halves.
// Per-lane elements: e<8 -> K = 8*hi + e ; e>=8 -> K = 16 + 8*hi + (e-8).
__device__ __forceinline__ v16bf load_a_frag(const unsigned short* __restrict__ row,
                                             int chunk, int hi) {
  ABu f;
  v8u p0 = *(const v8u*)(row + chunk + 8 * hi);
  v8u p1 = *(const v8u*)(row + chunk + 8 * hi + 16);
#pragma unroll
  for (int e = 0; e < 8; ++e) { f.u[e] = p0[e]; f.u[8 + e] = p1[e]; }
  return f.b;
}

// B fragment (32x16 bf16): lane = N (%16), element e = K in the same order as A.
__device__ __forceinline__ v16bf load_b_frag(const unsigned short* __restrict__ w,
                                             int kd, int ldn, int n0) {
  ABu f;
  int l = kd & 31, c0 = kd - l, bh = l >> 4;
#pragma unroll
  for (int e = 0; e < 16; ++e) {
    int k = c0 + (e < 8 ? 8 * bh + e : 16 + 8 * bh + (e - 8));
    f.u[e] = w[(size_t)k * ldn + n0 + (l & 15)];
  }
  return f.b;
}

__device__ __forceinline__ v8f wmma_bf16(v16bf a, v16bf b, v8f c) {
  return __builtin_amdgcn_wmma_f32_16x16x32_bf16(false, a, false, b, (short)0, c,
                                                 false, false);
}

// ---------------------------------------------------------------- conversion
__global__ void cvt_f32_bf16(const float* __restrict__ src,
                             unsigned short* __restrict__ dst, int n) {
  int i = blockIdx.x * blockDim.x + threadIdx.x;
  if (i < n) dst[i] = f2bf(src[i]);
}

// --------------------------------------- fc: x = relu(h@Wfc+b), 16x64 / wave
__global__ void __launch_bounds__(128)
fc_gemm(const unsigned short* __restrict__ hbf, const unsigned short* __restrict__ wbf,
        const float* __restrict__ bias, unsigned short* __restrict__ xbf) {
  int wave = threadIdx.x >> 5, lane = threadIdx.x & 31;
  int tile = blockIdx.x * 4 + wave;          // 1024 row-tiles * 8 col-groups
  int rowTile = tile >> 3, colGrp = tile & 7;
  int row0 = rowTile * 16, h0 = colGrp * 64;
  int hi = lane >> 4, m = lane & 15;
  const unsigned short* arow = hbf + (size_t)(row0 + m) * DIN;
  v8f c[4] = {};
  for (int ch = 0; ch < DIN; ch += 32) {
    v16bf a = load_a_frag(arow, ch, hi);
#pragma unroll
    for (int t = 0; t < 4; ++t) {
      v16bf b = load_b_frag(wbf, ch + lane, HID, h0 + 16 * t);
      c[t] = wmma_bf16(a, b, c[t]);
    }
  }
#pragma unroll
  for (int t = 0; t < 4; ++t) {
    float bn = bias[h0 + 16 * t + m];
#pragma unroll
    for (int r = 0; r < 8; ++r) {
      int orow = row0 + r + 8 * hi;
      float v = c[t][r] + bn;
      v = v > 0.f ? v : 0.f;
      xbf[(size_t)orow * HID + h0 + 16 * t + m] = f2bf(v);
    }
  }
}

// ---------------------------------------------------------------- ||x||^2
__global__ void __launch_bounds__(256)
sqnorm_k(const unsigned short* __restrict__ xbf, float* __restrict__ sq) {
  int wave = threadIdx.x >> 5, lane = threadIdx.x & 31;
  int row = blockIdx.x * 8 + wave;
  const unsigned short* p = xbf + (size_t)row * HID;
  float s = 0.f;
  for (int e = lane; e < HID; e += 32) { float v = bf2f(p[e]); s += v * v; }
#pragma unroll
  for (int off = 16; off; off >>= 1) s += __shfl_xor(s, off, 32);
  if (lane == 0) sq[row] = s;
}

// --------------------------------------------- kNN via WMMA Gram + top-5 keep
__global__ void __launch_bounds__(32)
knn_k(const unsigned short* __restrict__ xbf, const float* __restrict__ sq,
      int* __restrict__ nbr) {
  __shared__ unsigned short colbuf[16 * HID];   // 16 KB column tile
  __shared__ float mval[16 * 16 * KNNK];
  __shared__ int   mid [16 * 16 * KNNK];
  int lane = threadIdx.x;
  int tile = blockIdx.x;                         // 1024 tiles, 128 per bag
  int bag = tile >> 7;
  int row0 = tile * 16;                          // global row base
  int hi = lane >> 4, m = lane & 15;
  const unsigned short* arow = xbf + (size_t)(row0 + m) * HID;

  float sqi[8];
#pragma unroll
  for (int r = 0; r < 8; ++r) sqi[r] = sq[row0 + r + 8 * hi];

  float val[8][KNNK]; int idl[8][KNNK];
#pragma unroll
  for (int r = 0; r < 8; ++r)
#pragma unroll
    for (int s = 0; s < KNNK; ++s) { val[r][s] = 3.4e38f; idl[r][s] = 0; }

  for (int ct = 0; ct < NPTS / 16; ++ct) {
    int col0 = ct * 16;                          // within-bag column base
    __syncthreads();                             // colbuf reuse guard
    {
      const char* srcb = (const char*)(xbf + ((size_t)bag * NPTS + col0) * HID);
      char* dstb = (char*)colbuf;
#if __has_builtin(__builtin_amdgcn_global_load_async_to_lds_b128)
      for (int t = lane; t < 1024; t += 32) {
        __builtin_amdgcn_global_load_async_to_lds_b128(
            (__attribute__((address_space(1))) v4i_t*)(srcb + (size_t)t * 16),
            (__attribute__((address_space(3))) v4i_t*)(dstb + (size_t)t * 16),
            0, 0);
      }
#if __has_builtin(__builtin_amdgcn_s_wait_asynccnt)
      __builtin_amdgcn_s_wait_asynccnt(0);
#else
      asm volatile("s_wait_asynccnt 0" ::: "memory");
#endif
#else
      const uint4* src = (const uint4*)srcb;
      uint4* dst = (uint4*)dstb;
      for (int t = lane; t < 1024; t += 32) dst[t] = src[t];
#endif
    }
    __syncthreads();

    v8f c = {};
    for (int ch = 0; ch < HID; ch += 32) {
      v16bf a = load_a_frag(arow, ch, hi);
      ABu bf;
      int kd = ch + lane;
#pragma unroll
      for (int e = 0; e < 16; ++e) bf.u[e] = colbuf[(size_t)m * HID + ch + (e < 8 ? 8 * hi + e : 16 + 8 * hi + (e - 8))];
      c = wmma_bf16(a, bf.b, c);
    }
    int j = col0 + m;
    float sqj = sq[(size_t)bag * NPTS + j];
#pragma unroll
    for (int r = 0; r < 8; ++r) {
      float dv = sqi[r] + sqj - 2.f * c[r];
      int dj = j;
#pragma unroll
      for (int s = 0; s < KNNK; ++s) {
        if (dv < val[r][s]) {
          float tv = val[r][s]; int ti = idl[r][s];
          val[r][s] = dv; idl[r][s] = dj; dv = tv; dj = ti;
        }
      }
    }
  }
  // merge 16 lanes x top5 per row
#pragma unroll
  for (int r = 0; r < 8; ++r) {
    int row = r + 8 * hi;
#pragma unroll
    for (int s = 0; s < KNNK; ++s) {
      mval[(row * 16 + m) * KNNK + s] = val[r][s];
      mid [(row * 16 + m) * KNNK + s] = idl[r][s];
    }
  }
  __syncthreads();
  if (lane < 16) {
    int row = lane;
    float* mv = mval + row * 16 * KNNK;
    int*   mi = mid  + row * 16 * KNNK;
    for (int t = 0; t < KNNK; ++t) {
      float best = 3.4e38f; int bi = 0, bq = 0;
      for (int q = 0; q < 16 * KNNK; ++q) {
        float v = mv[q];
        if (v < best) { best = v; bi = mi[q]; bq = q; }
      }
      mv[bq] = 3.4e38f;
      nbr[(size_t)(row0 + row) * KNNK + t] = bi;   // within-bag index
    }
  }
}

// -------- edge conv: g = max_k relu([xi|xj-xi]@Wg + b), 16x64 tile per wave
__global__ void __launch_bounds__(128)
edgeconv_k(const unsigned short* __restrict__ xbf, const int* __restrict__ nbr,
           const unsigned short* __restrict__ wgbf, const float* __restrict__ bgv,
           float* __restrict__ gout, unsigned short* __restrict__ gbf) {
  int wave = threadIdx.x >> 5, lane = threadIdx.x & 31;
  int tile = blockIdx.x * 4 + wave;          // 1024 row-tiles * 8 h-groups
  int rowTile = tile >> 3, hGrp = tile & 7;
  int row0 = rowTile * 16, h0 = hGrp * 64;
  int bag = row0 >> 11;
  int hi = lane >> 4, m = lane & 15;
  int myrow = row0 + m;
  const unsigned short* xi = xbf + (size_t)myrow * HID;
  float gmax[4][8];
#pragma unroll
  for (int t = 0; t < 4; ++t)
#pragma unroll
    for (int r = 0; r < 8; ++r) gmax[t][r] = 0.f;   // relu output >= 0
  float bn[4];
#pragma unroll
  for (int t = 0; t < 4; ++t) bn[t] = bgv[h0 + 16 * t + m];

  for (int k = 0; k < KNNK; ++k) {
    int jn = nbr[(size_t)myrow * KNNK + k];
    const unsigned short* xj = xbf + ((size_t)bag * NPTS + jn) * HID;
    v8f c[4] = {};
    for (int ch = 0; ch < 2 * HID; ch += 32) {
      ABu a;
      if (ch < HID) {                          // xi half
        v8u p0 = *(const v8u*)(xi + ch + 8 * hi);
        v8u p1 = *(const v8u*)(xi + ch + 8 * hi + 16);
#pragma unroll
        for (int e = 0; e < 8; ++e) { a.u[e] = p0[e]; a.u[8 + e] = p1[e]; }
      } else {                                 // (xj - xi) half
        int cd = ch - HID;
        v8u pj0 = *(const v8u*)(xj + cd + 8 * hi);
        v8u pj1 = *(const v8u*)(xj + cd + 8 * hi + 16);
        v8u pi0 = *(const v8u*)(xi + cd + 8 * hi);
        v8u pi1 = *(const v8u*)(xi + cd + 8 * hi + 16);
#pragma unroll
        for (int e = 0; e < 8; ++e) {
          a.u[e]     = f2bf(bf2f(pj0[e]) - bf2f(pi0[e]));
          a.u[8 + e] = f2bf(bf2f(pj1[e]) - bf2f(pi1[e]));
        }
      }
#pragma unroll
      for (int t = 0; t < 4; ++t) {
        v16bf b = load_b_frag(wgbf, ch + lane, HID, h0 + 16 * t);
        c[t] = wmma_bf16(a.b, b, c[t]);
      }
    }
#pragma unroll
    for (int t = 0; t < 4; ++t)
#pragma unroll
      for (int r = 0; r < 8; ++r) {
        float v = c[t][r] + bn[t];
        v = v > 0.f ? v : 0.f;
        gmax[t][r] = gmax[t][r] > v ? gmax[t][r] : v;
      }
  }
#pragma unroll
  for (int t = 0; t < 4; ++t)
#pragma unroll
    for (int r = 0; r < 8; ++r) {
      size_t off = (size_t)(row0 + r + 8 * hi) * HID + h0 + 16 * t + m;
      gout[off] = gmax[t][r];
      gbf[off]  = f2bf(gmax[t][r]);
    }
}

// ------------------------------------------------ A_raw init with bias b_c
__global__ void init_araw(float* __restrict__ araw, const float* __restrict__ bc) {
  int i = blockIdx.x * blockDim.x + threadIdx.x;
  if (i < MROWS) araw[i] = bc[0];
}

// ----- attention: A_raw += sum_da tanh(..)*sigmoid(..)*W_c[da], 16x64 / wave
__global__ void __launch_bounds__(128)
attn_k(const unsigned short* __restrict__ gbf,
       const unsigned short* __restrict__ wabf, const float* __restrict__ ba,
       const unsigned short* __restrict__ wbbf, const float* __restrict__ bb,
       const float* __restrict__ wc, float* __restrict__ araw) {
  int wave = threadIdx.x >> 5, lane = threadIdx.x & 31;
  int tile = blockIdx.x * 4 + wave;          // 1024 row-tiles * 4 da-groups
  int rowTile = tile >> 2, daGrp = tile & 3;
  int row0 = rowTile * 16, d0 = daGrp * 64;
  int hi = lane >> 4, m = lane & 15;
  const unsigned short* grow = gbf + (size_t)(row0 + m) * HID;
  v8f ca[4] = {}, cb[4] = {};
  for (int ch = 0; ch < HID; ch += 32) {
    v16bf a = load_a_frag(grow, ch, hi);
#pragma unroll
    for (int t = 0; t < 4; ++t) {
      v16bf b1 = load_b_frag(wabf, ch + lane, DATT, d0 + 16 * t);
      v16bf b2 = load_b_frag(wbbf, ch + lane, DATT, d0 + 16 * t);
      ca[t] = wmma_bf16(a, b1, ca[t]);
      cb[t] = wmma_bf16(a, b2, cb[t]);
    }
  }
  float srow[8] = {0.f, 0.f, 0.f, 0.f, 0.f, 0.f, 0.f, 0.f};
#pragma unroll
  for (int t = 0; t < 4; ++t) {
    float wcn = wc[d0 + 16 * t + m];
    float ban = ba[d0 + 16 * t + m], bbn = bb[d0 + 16 * t + m];
#pragma unroll
    for (int r = 0; r < 8; ++r) {
      float av = tanhf(ca[t][r] + ban);
      float gv = 1.f / (1.f + __expf(-(cb[t][r] + bbn)));
      srow[r] += av * gv * wcn;
    }
  }
#pragma unroll
  for (int r = 0; r < 8; ++r) {
    float s = srow[r];
    s += __shfl_xor(s, 1, 32);
    s += __shfl_xor(s, 2, 32);
    s += __shfl_xor(s, 4, 32);
    s += __shfl_xor(s, 8, 32);
    if (m == 0) atomicAdd(&araw[row0 + r + 8 * hi], s);
  }
}

// ------------------- per-bag softmax, weighted pooling, logits, Y_prob
__global__ void __launch_bounds__(256)
finalize_k(const float* __restrict__ araw, const float* __restrict__ g,
           const float* __restrict__ wcls, const float* __restrict__ bcls,
           float* __restrict__ logits, float* __restrict__ yprob) {
  __shared__ float red[256];
  __shared__ float w[NPTS];
  __shared__ float Mv[HID];
  __shared__ float lsh[NCLS];
  int bag = blockIdx.x, tid = threadIdx.x;
  const float* ab = araw + (size_t)bag * NPTS;

  float mx = -3.4e38f;
  for (int i = tid; i < NPTS; i += 256) mx = fmaxf(mx, ab[i]);
  red[tid] = mx; __syncthreads();
  for (int s = 128; s; s >>= 1) { if (tid < s) red[tid] = fmaxf(red[tid], red[tid + s]); __syncthreads(); }
  mx = red[0]; __syncthreads();

  float sum = 0.f;
  for (int i = tid; i < NPTS; i += 256) { float e = __expf(ab[i] - mx); w[i] = e; sum += e; }
  red[tid] = sum; __syncthreads();
  for (int s = 128; s; s >>= 1) { if (tid < s) red[tid] += red[tid + s]; __syncthreads(); }
  float inv = 1.f / red[0];
  __syncthreads();

  for (int h = tid; h < HID; h += 256) {
    float acc = 0.f;
    const float* gp = g + (size_t)bag * NPTS * HID + h;
    for (int n = 0; n < NPTS; ++n) acc += w[n] * gp[(size_t)n * HID];
    Mv[h] = acc * inv;
  }
  __syncthreads();

  if (tid < NCLS) {
    float acc = bcls[tid];
    for (int h2 = 0; h2 < HID; ++h2) acc += Mv[h2] * wcls[h2 * NCLS + tid];
    logits[bag * NCLS + tid] = acc;
    lsh[tid] = acc;
  }
  __syncthreads();
  if (tid < NCLS) {
    float mm = fmaxf(lsh[0], lsh[1]);
    float e0 = __expf(lsh[0] - mm), e1 = __expf(lsh[1] - mm);
    yprob[bag * NCLS + tid] = (tid == 0 ? e0 : e1) / (e0 + e1);
  }
}

extern "C" void kernel_launch(void* const* d_in, const int* in_sizes, int n_in,
                              void* d_out, int out_size, void* d_ws, size_t ws_size,
                              hipStream_t stream) {
  (void)in_sizes; (void)n_in; (void)out_size; (void)ws_size;
  const float* h    = (const float*)d_in[0];
  const float* Wfc  = (const float*)d_in[1];
  const float* bfc  = (const float*)d_in[2];
  const float* Wg   = (const float*)d_in[3];
  const float* bgv  = (const float*)d_in[4];
  const float* Wa   = (const float*)d_in[5];
  const float* ba   = (const float*)d_in[6];
  const float* Wb   = (const float*)d_in[7];
  const float* bb   = (const float*)d_in[8];
  const float* Wc   = (const float*)d_in[9];
  const float* bc   = (const float*)d_in[10];
  const float* Wcls = (const float*)d_in[11];
  const float* bcls = (const float*)d_in[12];

  float* out    = (float*)d_out;
  float* logits = out;            // (8,2)
  float* yprob  = out + 16;       // (8,2)
  float* araw   = out + 32;       // (8,2048,1)

  char* ws = (char*)d_ws;
  size_t off = 0;
  auto alloc = [&](size_t bytes) -> char* {
    char* p = ws + off; off += (bytes + 255) & ~(size_t)255; return p;
  };
  unsigned short* hbf  = (unsigned short*)alloc((size_t)BAGS * NPTS * DIN * 2);
  unsigned short* wfcb = (unsigned short*)alloc((size_t)DIN * HID * 2);
  unsigned short* wgb  = (unsigned short*)alloc((size_t)2 * HID * HID * 2);
  unsigned short* wab  = (unsigned short*)alloc((size_t)HID * DATT * 2);
  unsigned short* wbb  = (unsigned short*)alloc((size_t)HID * DATT * 2);
  unsigned short* xbf  = (unsigned short*)alloc((size_t)MROWS * HID * 2);
  float*          sq   = (float*)alloc((size_t)MROWS * 4);
  int*            nbr  = (int*)alloc((size_t)MROWS * KNNK * 4);
  unsigned short* gbf  = (unsigned short*)alloc((size_t)MROWS * HID * 2);
  float*          gf   = (float*)alloc((size_t)MROWS * HID * 4);

  int nh = BAGS * NPTS * DIN;
  cvt_f32_bf16<<<(nh + 255) / 256, 256, 0, stream>>>(h, hbf, nh);
  cvt_f32_bf16<<<(DIN * HID + 255) / 256, 256, 0, stream>>>(Wfc, wfcb, DIN * HID);
  cvt_f32_bf16<<<(2 * HID * HID + 255) / 256, 256, 0, stream>>>(Wg, wgb, 2 * HID * HID);
  cvt_f32_bf16<<<(HID * DATT + 255) / 256, 256, 0, stream>>>(Wa, wab, HID * DATT);
  cvt_f32_bf16<<<(HID * DATT + 255) / 256, 256, 0, stream>>>(Wb, wbb, HID * DATT);

  fc_gemm   <<<2048, 128, 0, stream>>>(hbf, wfcb, bfc, xbf);
  sqnorm_k  <<<2048, 256, 0, stream>>>(xbf, sq);
  knn_k     <<<1024,  32, 0, stream>>>(xbf, sq, nbr);
  edgeconv_k<<<2048, 128, 0, stream>>>(xbf, nbr, wgb, bgv, gf, gbf);
  init_araw <<<64,   256, 0, stream>>>(araw, bc);
  attn_k    <<<1024, 128, 0, stream>>>(gbf, wab, ba, wbb, bb, Wc, araw);
  finalize_k<<<BAGS, 256, 0, stream>>>(araw, gf, Wcls, bcls, logits, yprob);
}
